// AttentionBlock_78743930405542
// MI455X (gfx1250) — matrix operand order repaired and checked by reading, not compiled, not run on hardware
//
#include <hip/hip_runtime.h>
#include <hip/hip_bf16.h>

// ---------------------------------------------------------------------------
// AttentionBlock for MI455X (gfx1250, wave32, WMMA 16x16x32 bf16)
// B=8, C=D=512, H=W=64, N=4096, 32 groups
// GEMMs use 4x2 register tiling (64x32 output per wave): 8 WMMA per 6 KB of
// L2 loads, ~140 VGPRs -> no scratch spills.
// ---------------------------------------------------------------------------

typedef __attribute__((ext_vector_type(16))) __bf16 v16bf;
typedef __attribute__((ext_vector_type(8)))  float  v8f;
typedef __attribute__((ext_vector_type(4)))  unsigned int u32x4;

#define Bv   8
#define Cv   512
#define Dv   512
#define Nv   4096
#define GRP  32

static __device__ inline v8f wmma_bf16(v16bf a, v16bf b, v8f c) {
  // D = A(16x32 bf16) * B(32x16 bf16) + C(16x16 f32)
  return __builtin_amdgcn_wmma_f32_16x16x32_bf16(
      /*neg_a=*/false, a, /*neg_b=*/false, b,
      /*c_mod=*/(short)0, c, /*reuse_a=*/false, /*reuse_b=*/false);
}

// A-matrix fragment (16x32, 16-bit): lane holds row M=lane%16.
// lanes 0-15: K=[0..7] and [16..23]; lanes 16-31: K=[8..15] and [24..31].
// Caller passes base already offset by (hi*8); second segment is +16 elements.
static __device__ inline v16bf load_frag_rowpair(const __bf16* base) {
  union { v16bf v; u32x4 u[2]; } f;
  f.u[0] = *(const u32x4*)(base);
  f.u[1] = *(const u32x4*)(base + 16);
  return f.v;
}

// B-matrix fragment (32x16, 16-bit): lane holds column N=lane%16,
// lanes 0-15: K=[0..7] and [16..23]; lanes 16-31: K=[8..15] and [24..31].
static __device__ inline v16bf load_frag_contig(const __bf16* base) {
  union { v16bf v; u32x4 u[2]; } f;
  f.u[0] = *(const u32x4*)(base);
  f.u[1] = *(const u32x4*)(base + 16);
  return f.v;
}

// ---------------------------------------------------------------------------
// Kernel 1: weight conversion fp32 -> bf16 (q_w | k_w | v_w | proj_w packed)
// ---------------------------------------------------------------------------
__global__ void wcvt_kernel(const float* __restrict__ qw, const float* __restrict__ kw,
                            const float* __restrict__ vw, const float* __restrict__ pw,
                            __bf16* __restrict__ dst) {
  int i = blockIdx.x * 256 + threadIdx.x;          // 4 * 512*512 total
  int t = i >> 18;
  int r = i & ((1 << 18) - 1);
  const float* s = (t == 0) ? qw : (t == 1) ? kw : (t == 2) ? vw : pw;
  dst[i] = (__bf16)s[r];
}

// ---------------------------------------------------------------------------
// Kernel 2: GroupNorm.  x (B,C,N) fp32 -> h (B,N,C) bf16
// One block per (batch, group): 16 channels x 4096 spatial = 65536 elements.
// ---------------------------------------------------------------------------
__global__ void gn_kernel(const float* __restrict__ x, const float* __restrict__ gamma,
                          const float* __restrict__ beta, __bf16* __restrict__ h) {
  const int b  = blockIdx.x >> 5;
  const int g  = blockIdx.x & 31;
  const int c0 = g * 16;
  const int tid = threadIdx.x;
  const float* xb = x + ((size_t)b * Cv + c0) * Nv;   // 16 contiguous channels

  float s = 0.f, s2 = 0.f;
  for (int i = tid; i < 16 * Nv; i += 256) {
    float v = xb[i];
    s += v; s2 += v * v;
  }
#pragma unroll
  for (int m = 16; m >= 1; m >>= 1) {
    s  += __shfl_xor(s,  m, 32);
    s2 += __shfl_xor(s2, m, 32);
  }
  __shared__ float rs[8], rs2[8];
  __shared__ float smean, sinv;
  if ((tid & 31) == 0) { rs[tid >> 5] = s; rs2[tid >> 5] = s2; }
  __syncthreads();
  if (tid == 0) {
    float S = 0.f, S2 = 0.f;
    for (int j = 0; j < 8; ++j) { S += rs[j]; S2 += rs2[j]; }
    float mean = S * (1.f / 65536.f);
    float var  = S2 * (1.f / 65536.f) - mean * mean;
    smean = mean;
    sinv  = rsqrtf(var + 1e-5f);
  }
  __syncthreads();
  const float mean = smean, inv = sinv;
  for (int i = tid; i < 16 * Nv; i += 256) {
    int cl = i >> 12, n = i & (Nv - 1);
    int c  = c0 + cl;
    float v = (xb[i] - mean) * inv * gamma[c] + beta[c];
    h[((size_t)b * Nv + n) * Cv + c] = (__bf16)v;
  }
}

// ---------------------------------------------------------------------------
// Kernel 3: QKV projections via WMMA, 4x2 register tiling (64x32 per wave).
//   q,k: out(n,d) = sum_c h(n,c) * W(d,c)   -> stored (B,N,D) bf16
//   v:   out(d,n) = sum_c W(d,c) * h(n,c)   -> stored (B,D,N) bf16
// q gets (acc + bias) * C^-0.5 folded in.
// Waves per (b,type): 64 row-groups x 16 col-groups = 1024.
// Total waves = 8*3*1024 = 24576 -> 3072 blocks of 8 waves.
// ---------------------------------------------------------------------------
__global__ void __launch_bounds__(256, 1)
qkv_kernel(const __bf16* __restrict__ h,
           const __bf16* __restrict__ wq, const __bf16* __restrict__ wk,
           const __bf16* __restrict__ wv,
           const float* __restrict__ qb, const float* __restrict__ kb,
           const float* __restrict__ vb,
           __bf16* __restrict__ q, __bf16* __restrict__ k,
           __bf16* __restrict__ v, float scale) {
  const int wid   = blockIdx.x * 8 + (threadIdx.x >> 5);
  const int lane  = threadIdx.x & 31;
  const int laneq = lane & 15;
  const int hi    = lane >> 4;

  const int b   = wid / 3072;
  const int rem = wid % 3072;
  const int t   = rem / 1024;             // 0=q 1=k 2=v
  const int sub = rem % 1024;             // 64 row-groups x 16 col-groups

  v8f acc[4][2] = {};

  if (t < 2) {
    const int n0 = (sub >> 4) * 64;
    const int d0 = (sub & 15) * 32;
    const __bf16* wm   = (t == 0) ? wq : wk;
    const float*  bias = (t == 0) ? qb : kb;
    __bf16*       outp = (t == 0) ? q : k;

    const __bf16* arow[4];
    const __bf16* bcol[2];
#pragma unroll
    for (int i = 0; i < 4; ++i)
      arow[i] = h + ((size_t)(b * Nv + n0 + i * 16 + laneq)) * Cv + hi * 8;
#pragma unroll
    for (int j = 0; j < 2; ++j)
      bcol[j] = wm + (size_t)(d0 + j * 16 + laneq) * Cv + hi * 8;

    for (int ks = 0; ks < 16; ++ks) {
      v16bf a[4], bb[2];
#pragma unroll
      for (int i = 0; i < 4; ++i) a[i]  = load_frag_rowpair(arow[i] + ks * 32);
#pragma unroll
      for (int j = 0; j < 2; ++j) bb[j] = load_frag_contig(bcol[j] + ks * 32);
#pragma unroll
      for (int i = 0; i < 4; ++i)
#pragma unroll
        for (int j = 0; j < 2; ++j)
          acc[i][j] = wmma_bf16(a[i], bb[j], acc[i][j]);
    }
    const float sc = (t == 0) ? scale : 1.f;
#pragma unroll
    for (int j = 0; j < 2; ++j) {
      const int d  = d0 + j * 16 + laneq;
      const float bv = bias[d];
#pragma unroll
      for (int i = 0; i < 4; ++i)
#pragma unroll
        for (int r = 0; r < 8; ++r) {
          int n = n0 + i * 16 + r + hi * 8;              // C-layout rows r / 8+r
          outp[((size_t)(b * Nv + n)) * Dv + d] = (__bf16)((acc[i][j][r] + bv) * sc);
        }
    }
  } else {
    const int n0 = (sub >> 4) * 64;      // 4 B-subtiles of 16 tokens
    const int d0 = (sub & 15) * 32;      // 2 A-subtiles of 16 channels
    const __bf16* arow[2];
    const __bf16* bcol[4];
#pragma unroll
    for (int i = 0; i < 2; ++i)
      arow[i] = wv + (size_t)(d0 + i * 16 + laneq) * Cv + hi * 8;
#pragma unroll
    for (int j = 0; j < 4; ++j)
      bcol[j] = h + ((size_t)(b * Nv + n0 + j * 16 + laneq)) * Cv + hi * 8;

    for (int ks = 0; ks < 16; ++ks) {
      v16bf a[2], bb[4];
#pragma unroll
      for (int i = 0; i < 2; ++i) a[i]  = load_frag_rowpair(arow[i] + ks * 32);
#pragma unroll
      for (int j = 0; j < 4; ++j) bb[j] = load_frag_contig(bcol[j] + ks * 32);
#pragma unroll
      for (int j = 0; j < 4; ++j)
#pragma unroll
        for (int i = 0; i < 2; ++i)
          acc[j][i] = wmma_bf16(a[i], bb[j], acc[j][i]);   // acc[bj][ai]
    }
#pragma unroll
    for (int i = 0; i < 2; ++i)
#pragma unroll
      for (int r = 0; r < 8; ++r) {
        int d = d0 + i * 16 + r + hi * 8;
        float bvv = vb[d];
#pragma unroll
        for (int j = 0; j < 4; ++j)
          v[((size_t)(b * Dv + d)) * Nv + n0 + j * 16 + laneq] =
              (__bf16)(acc[j][i][r] + bvv);
      }
  }
}

// ---------------------------------------------------------------------------
// Kernel 4: attention for one 16-token row block per workgroup (8 waves).
// Phase 1: S(16x4096) = Q Kt (pre-scaled) via WMMA -> LDS bf16 (128 KB)
// Phase 2: exact row softmax in LDS
// Phase 3: O(16x512) = P V via WMMA (A-fragments from LDS) -> hf (B,N,D) bf16
// ---------------------------------------------------------------------------
__global__ void __launch_bounds__(256, 1)
attn_kernel(const __bf16* __restrict__ q, const __bf16* __restrict__ k,
            const __bf16* __restrict__ v, __bf16* __restrict__ hf) {
  extern __shared__ __bf16 P[];            // 16 x 4096 bf16 = 128 KB
  __shared__ float sred[256];

  const int b    = blockIdx.x >> 8;
  const int nt   = blockIdx.x & 255;
  const int n0   = nt * 16;
  const int warp = threadIdx.x >> 5;
  const int lane = threadIdx.x & 31;
  const int laneq = lane & 15;
  const int hi    = lane >> 4;

  // ---- Phase 1: S row-block into LDS; Q fragments stay resident ----
  v16bf qf[16];
  {
    const __bf16* qrow = q + ((size_t)(b * Nv + n0 + laneq)) * Dv + hi * 8;
#pragma unroll
    for (int ks = 0; ks < 16; ++ks) qf[ks] = load_frag_rowpair(qrow + ks * 32);
  }
  for (int mtl = 0; mtl < 32; ++mtl) {
    const int m0 = (warp * 32 + mtl) * 16;
    const __bf16* kcol = k + ((size_t)(b * Nv + m0 + laneq)) * Dv + hi * 8;
    v8f acc = {};
#pragma unroll
    for (int ks = 0; ks < 16; ++ks) {
      v16bf bb = load_frag_contig(kcol + ks * 32);
      acc = wmma_bf16(qf[ks], bb, acc);
    }
#pragma unroll
    for (int r = 0; r < 8; ++r)
      P[(r + hi * 8) * Nv + m0 + laneq] = (__bf16)acc[r];
  }
  __syncthreads();

  // ---- Phase 2: softmax over each of the 16 rows (4096 cols) ----
  {
    const int row  = threadIdx.x >> 4;     // 0..15
    const int tcol = threadIdx.x & 15;     // 16 threads per row, 256 cols each
    __bf16* prow = P + row * Nv;
    float lmax = -3.402823466e38f;
    for (int c2 = tcol * 256; c2 < tcol * 256 + 256; ++c2)
      lmax = fmaxf(lmax, (float)prow[c2]);
    sred[threadIdx.x] = lmax;
    __syncthreads();
    float rmax = -3.402823466e38f;
    for (int j = 0; j < 16; ++j) rmax = fmaxf(rmax, sred[(row << 4) + j]);
    __syncthreads();
    float lsum = 0.f;
    for (int c2 = tcol * 256; c2 < tcol * 256 + 256; ++c2) {
      float e = __expf((float)prow[c2] - rmax);
      prow[c2] = (__bf16)e;
      lsum += e;
    }
    sred[threadIdx.x] = lsum;
    __syncthreads();
    float rsum = 0.f;
    for (int j = 0; j < 16; ++j) rsum += sred[(row << 4) + j];
    const float invs = 1.f / rsum;
    for (int c2 = tcol * 256; c2 < tcol * 256 + 256; ++c2)
      prow[c2] = (__bf16)((float)prow[c2] * invs);
  }
  __syncthreads();

  // ---- Phase 3: O = P * Vt ; each wave owns 64 d-columns ----
  const int d0 = warp * 64;
  v8f oacc[4] = {{}, {}, {}, {}};
  for (int ms = 0; ms < 128; ++ms) {
    const __bf16* prow = P + laneq * Nv + ms * 32 + hi * 8;
    union { v16bf vv; u32x4 u[2]; } af;
    af.u[0] = *(const u32x4*)(prow);
    af.u[1] = *(const u32x4*)(prow + 16);
#pragma unroll
    for (int j = 0; j < 4; ++j) {
      const __bf16* vcol =
          v + ((size_t)(b * Dv + d0 + j * 16 + laneq)) * Nv + ms * 32 + hi * 8;
      v16bf bb = load_frag_contig(vcol);
      oacc[j] = wmma_bf16(af.vv, bb, oacc[j]);
    }
  }
#pragma unroll
  for (int j = 0; j < 4; ++j)
#pragma unroll
    for (int r = 0; r < 8; ++r) {
      int n = n0 + r + hi * 8;
      int d = d0 + j * 16 + laneq;
      hf[((size_t)(b * Nv + n)) * Dv + d] = (__bf16)oacc[j][r];
    }
}

// ---------------------------------------------------------------------------
// Kernel 5: out(b,c,n) = x(b,c,n) + proj_b[c] + sum_d hf(b,n,d) * proj_w(c,d)
// 4x2 register tiling: 64 tokens x 32 channels per wave.
// Waves per batch: 64 x 16 = 1024; blocks = B*1024/8 = 1024.
// ---------------------------------------------------------------------------
__global__ void __launch_bounds__(256, 1)
proj_kernel(const __bf16* __restrict__ hf, const __bf16* __restrict__ wp,
            const float* __restrict__ pb, const float* __restrict__ x,
            float* __restrict__ out) {
  const int wid   = blockIdx.x * 8 + (threadIdx.x >> 5);
  const int lane  = threadIdx.x & 31;
  const int laneq = lane & 15;
  const int hi    = lane >> 4;
  const int b   = wid >> 10;
  const int sub = wid & 1023;
  const int n0 = (sub >> 4) * 64;
  const int c0 = (sub & 15) * 32;

  const __bf16* arow[4];
  const __bf16* bcol[2];
#pragma unroll
  for (int i = 0; i < 4; ++i)
    arow[i] = hf + ((size_t)(b * Nv + n0 + i * 16 + laneq)) * Dv + hi * 8;
#pragma unroll
  for (int j = 0; j < 2; ++j)
    bcol[j] = wp + (size_t)(c0 + j * 16 + laneq) * Dv + hi * 8;

  v8f acc[4][2] = {};
  for (int ks = 0; ks < 16; ++ks) {
    v16bf a[4], bb[2];
#pragma unroll
    for (int i = 0; i < 4; ++i) a[i]  = load_frag_rowpair(arow[i] + ks * 32);
#pragma unroll
    for (int j = 0; j < 2; ++j) bb[j] = load_frag_contig(bcol[j] + ks * 32);
#pragma unroll
    for (int i = 0; i < 4; ++i)
#pragma unroll
      for (int j = 0; j < 2; ++j)
        acc[i][j] = wmma_bf16(a[i], bb[j], acc[i][j]);
  }
#pragma unroll
  for (int j = 0; j < 2; ++j) {
    const int c = c0 + j * 16 + laneq;
    const float bias = pb[c];
#pragma unroll
    for (int i = 0; i < 4; ++i)
#pragma unroll
      for (int r = 0; r < 8; ++r) {
        int n = n0 + i * 16 + r + hi * 8;
        size_t idx = ((size_t)b * Cv + c) * Nv + n;
        out[idx] = x[idx] + acc[i][j][r] + bias;
      }
  }
}

// ---------------------------------------------------------------------------
extern "C" void kernel_launch(void* const* d_in, const int* in_sizes, int n_in,
                              void* d_out, int out_size, void* d_ws, size_t ws_size,
                              hipStream_t stream) {
  (void)in_sizes; (void)n_in; (void)out_size; (void)ws_size;
  const float* x    = (const float*)d_in[0];
  const float* gn_w = (const float*)d_in[1];
  const float* gn_b = (const float*)d_in[2];
  const float* q_w  = (const float*)d_in[3];
  const float* q_b  = (const float*)d_in[4];
  const float* k_w  = (const float*)d_in[5];
  const float* k_b  = (const float*)d_in[6];
  const float* v_w  = (const float*)d_in[7];
  const float* v_b  = (const float*)d_in[8];
  const float* p_w  = (const float*)d_in[9];
  const float* p_b  = (const float*)d_in[10];
  float* out = (float*)d_out;

  char* ws = (char*)d_ws;
  size_t off = 0;
  auto alloc = [&](size_t bytes) -> void* {
    void* p = ws + off;
    off += (bytes + 255) & ~(size_t)255;
    return p;
  };
  const size_t bnc = (size_t)Bv * Nv * Cv;            // 16.8M elements
  __bf16* h    = (__bf16*)alloc(bnc * 2);             // (B,N,C)
  __bf16* q    = (__bf16*)alloc(bnc * 2);             // (B,N,D)
  __bf16* k    = (__bf16*)alloc(bnc * 2);             // (B,N,D)
  __bf16* v    = (__bf16*)alloc(bnc * 2);             // (B,D,N)
  __bf16* hf   = (__bf16*)alloc(bnc * 2);             // (B,N,D)
  __bf16* wbf  = (__bf16*)alloc((size_t)4 * Cv * Dv * 2);  // wq|wk|wv|wp
  __bf16* wq = wbf;
  __bf16* wk = wbf + (size_t)Cv * Dv;
  __bf16* wv = wbf + (size_t)2 * Cv * Dv;
  __bf16* wp = wbf + (size_t)3 * Cv * Dv;

  const float scale = 0.044194173824159216f;          // 512^-0.5

  // 1. convert weights to bf16
  wcvt_kernel<<<(4 * Cv * Dv) / 256, 256, 0, stream>>>(q_w, k_w, v_w, p_w, wbf);
  // 2. group norm -> h (B,N,C) bf16
  gn_kernel<<<Bv * GRP, 256, 0, stream>>>(x, gn_w, gn_b, h);
  // 3. QKV projections: B*3072 waves, 8 per block
  qkv_kernel<<<3072, 256, 0, stream>>>(h, wq, wk, wv, q_b, k_b, v_b, q, k, v, scale);
  // 4. attention: one block per (batch, 16-token tile); 128 KB dynamic LDS
  attn_kernel<<<Bv * (Nv / 16), 256, 16 * Nv * sizeof(__bf16), stream>>>(q, k, v, hf);
  // 5. projection + residual
  proj_kernel<<<1024, 256, 0, stream>>>(hf, wp, p_b, x, out);
}
